// MORSpyMaster_82042465288708
// MI455X (gfx1250) — compile-verified, matrix operand order checked by reading, not run on hardware
//
#include <hip/hip_runtime.h>
#include <hip/hip_bf16.h>

typedef __bf16 bf16_t;
typedef __attribute__((ext_vector_type(16))) __bf16 v16bf;
typedef __attribute__((ext_vector_type(8)))  float  v8f;

#define NB   256      // batch
#define DD   768      // embedding dim
#define NV   100000   // vocab rows
#define NVP  100032   // vocab rows padded to 64
#define VK   80       // top-k words
#define LSTR 72       // padded LDS row stride (bf16 elems): 144B = 36 banks -> conflict-free

// ---------------- CDNA5 async global->LDS helpers ----------------
__device__ __forceinline__ unsigned lds_u32(void* p) {
  // generic -> LDS(AS3) pointer -> 32-bit LDS byte offset
  return (unsigned)(unsigned long long)(__attribute__((address_space(3))) char*)(p);
}
// Copy one thread's slice of an (A:64B, B:32B) tile using async DMA.
// INST_OFFSET applies to both LDS and global address (ISA 8.2), so all chunks
// share one address VGPR pair and use immediate offsets.
__device__ __forceinline__ void async_tile_cp(unsigned ldsA, const void* ga,
                                              unsigned ldsB, const void* gb) {
  asm volatile(
      "global_load_async_to_lds_b128 %0, %2, off\n\t"
      "global_load_async_to_lds_b128 %0, %2, off offset:16\n\t"
      "global_load_async_to_lds_b128 %0, %2, off offset:32\n\t"
      "global_load_async_to_lds_b128 %0, %2, off offset:48\n\t"
      "global_load_async_to_lds_b128 %1, %3, off\n\t"
      "global_load_async_to_lds_b128 %1, %3, off offset:16"
      :: "v"(ldsA), "v"(ldsB), "v"(ga), "v"(gb) : "memory");
}
__device__ __forceinline__ void wait_async0() {
  asm volatile("s_wait_asynccnt 0x0" ::: "memory");
}

// ---------------- block reduction helper ----------------
__device__ __forceinline__ float block_reduce_sum(float v, float* sbuf) {
  for (int o = 16; o > 0; o >>= 1) v += __shfl_down(v, o, 32);
  int lane = threadIdx.x & 31, wid = threadIdx.x >> 5;
  if (lane == 0) sbuf[wid] = v;
  __syncthreads();
  if (threadIdx.x == 0) {
    float s = 0.f;
    int nw = (blockDim.x + 31) >> 5;
    for (int i = 0; i < nw; ++i) s += sbuf[i];
    sbuf[8] = s;
  }
  __syncthreads();
  float out = sbuf[8];
  __syncthreads();
  return out;
}

// ---------------- pooling + per-word normalization ----------------
__device__ __forceinline__ void pool_group(const float* src, int nw, bf16_t* dst, float* red) {
  float m[3] = {0.f, 0.f, 0.f};
  for (int w = 0; w < nw; ++w)
    for (int q = 0; q < 3; ++q) m[q] += src[w * DD + threadIdx.x + q * 256];
  float ss = 0.f;
  for (int q = 0; q < 3; ++q) { m[q] /= (float)nw; ss += m[q] * m[q]; }
  ss = block_reduce_sum(ss, red);
  float inv = 1.f / fmaxf(sqrtf(ss), 1e-12f);
  for (int q = 0; q < 3; ++q) dst[threadIdx.x + q * 256] = (bf16_t)(m[q] * inv);
}

__global__ __launch_bounds__(256)
void k_pool(const float* __restrict__ pos, const float* __restrict__ neg,
            const float* __restrict__ neut, const float* __restrict__ assas,
            bf16_t* __restrict__ xb, float* __restrict__ board_n) {
  int b = blockIdx.x;
  __shared__ float red[9];
  pool_group(neg + (size_t)b * 9 * DD, 9, xb + (size_t)b * 3072 + 0, red);
  for (int q = 0; q < 3; ++q) {
    int d = threadIdx.x + q * 256;
    xb[(size_t)b * 3072 + DD + d] = (bf16_t)assas[(size_t)b * DD + d];
  }
  pool_group(neut + (size_t)b * 7 * DD, 7, xb + (size_t)b * 3072 + 2 * DD, red);
  pool_group(pos + (size_t)b * 9 * DD, 9, xb + (size_t)b * 3072 + 3 * DD, red);
  for (int j = 0; j < 26; ++j) {
    const float* src = (j < 9)  ? pos  + ((size_t)b * 9 + j) * DD
                     : (j < 18) ? neg  + ((size_t)b * 9 + (j - 9)) * DD
                     : (j < 25) ? neut + ((size_t)b * 7 + (j - 18)) * DD
                                : assas + (size_t)b * DD;
    float vv[3]; float ss = 0.f;
    for (int q = 0; q < 3; ++q) { vv[q] = src[threadIdx.x + q * 256]; ss += vv[q] * vv[q]; }
    ss = block_reduce_sum(ss, red);
    float inv = 1.f / fmaxf(sqrtf(ss), 1e-8f);
    for (int q = 0; q < 3; ++q)
      board_n[((size_t)b * 26 + j) * DD + threadIdx.x + q * 256] = vv[q] * inv;
  }
}

// ---------------- weight transpose + f32->bf16 (padded, zero-filled) ----------------
__global__ __launch_bounds__(256)
void k_wt_cvt(const float* __restrict__ W, bf16_t* __restrict__ Wt,
              int K, int N, int Kp, int Np) {
  __shared__ float tile[32][33];
  int kb = blockIdx.x * 32;
  int nb = blockIdx.y * 32;
  int tx = threadIdx.x & 31, ty = threadIdx.x >> 5;
  for (int i = ty; i < 32; i += 8) {
    int k = kb + i, n = nb + tx;
    tile[i][tx] = (k < K && n < N) ? W[(size_t)k * N + n] : 0.f;
  }
  __syncthreads();
  for (int i = ty; i < 32; i += 8) {
    int n = nb + i, k = kb + tx;
    if (n < Np && k < Kp) Wt[(size_t)n * Kp + k] = (bf16_t)tile[tx][i];
  }
}

// ---------------- vocab L2-normalize -> bf16 (padded rows zeroed) ----------------
__global__ __launch_bounds__(256)
void k_vocab(const float* __restrict__ vocab, bf16_t* __restrict__ vb, int nreal) {
  int v = blockIdx.x;
  __shared__ float red[9];
  if (v >= nreal) {
    for (int q = 0; q < 3; ++q) vb[(size_t)v * DD + threadIdx.x + q * 256] = (bf16_t)0.f;
    return;
  }
  const float* src = vocab + (size_t)v * DD;
  float vv[3]; float ss = 0.f;
  for (int q = 0; q < 3; ++q) { vv[q] = src[threadIdx.x + q * 256]; ss += vv[q] * vv[q]; }
  ss = block_reduce_sum(ss, red);
  float inv = 1.f / fmaxf(sqrtf(ss), 1e-12f);
  for (int q = 0; q < 3; ++q) vb[(size_t)v * DD + threadIdx.x + q * 256] = (bf16_t)(vv[q] * inv);
}

// ---------------- bf16 WMMA TN GEMM: C[M,N] = A[M,K] * Bt[N,K]^T ----------------
// block tile 128x64, 8 waves, K-step 64, double-buffered LDS filled by
// GLOBAL_LOAD_ASYNC_TO_LDS_B128 (ASYNCcnt). Row stride 72 bf16 (144B = 36
// banks) makes the 16-row b128 fragment loads LDS-bank-conflict-free.
__global__ __launch_bounds__(256)
void gemm_tn_bf16(const bf16_t* __restrict__ A, const bf16_t* __restrict__ Bt,
                  const float* __restrict__ bias,
                  float* __restrict__ Cf, bf16_t* __restrict__ Cb,
                  int M, int N, int K, int Nreal, int relu, float padfill) {
  __shared__ __align__(16) bf16_t As[2][128 * LSTR];  // 2 x 18KB
  __shared__ __align__(16) bf16_t Bs[2][64 * LSTR];   // 2 x 9KB
  const int n0 = blockIdx.x * 64;
  const int m0 = blockIdx.y * 128;
  const int tid = threadIdx.x;
  const int wave = tid >> 5, lane = tid & 31;
  const int wm = wave & 1;    // M half (64 rows)
  const int wn = wave >> 1;   // N quarter (16 cols)
  const int lrow = lane & 15, lhalf = lane >> 4;

  const int arow = tid >> 1, ahalf = tid & 1;   // A: 64B (32 bf16) per thread
  const int brow = tid >> 2, bq = tid & 3;      // B: 32B (16 bf16) per thread

  const bf16_t* gA = A + (size_t)(m0 + arow) * K + ahalf * 32;
  const bf16_t* gB = Bt + (size_t)(n0 + brow) * K + bq * 16;
  unsigned lA[2] = { lds_u32(&As[0][arow * LSTR + ahalf * 32]),
                     lds_u32(&As[1][arow * LSTR + ahalf * 32]) };
  unsigned lB[2] = { lds_u32(&Bs[0][brow * LSTR + bq * 16]),
                     lds_u32(&Bs[1][brow * LSTR + bq * 16]) };

  const v8f vzero = {0.f, 0.f, 0.f, 0.f, 0.f, 0.f, 0.f, 0.f};
  v8f acc0 = vzero, acc1 = vzero, acc2 = vzero, acc3 = vzero;
  const int T = K >> 6;

  // prologue: tile 0 -> buffer 0
  async_tile_cp(lA[0], gA, lB[0], gB);
  wait_async0();
  __syncthreads();

  for (int t = 0; t < T; ++t) {
    const int cur = t & 1, nxt = cur ^ 1;
    if (t + 1 < T)
      async_tile_cp(lA[nxt], gA + (size_t)(t + 1) * 64,
                    lB[nxt], gB + (size_t)(t + 1) * 64);

    const bf16_t* as = As[cur];
    const bf16_t* bs = Bs[cur];
    // load all fragments first so the 8 WMMAs issue back-to-back
    v16bf bf0 = *(const v16bf*)&bs[(wn * 16 + lrow) * LSTR +  0 + lhalf * 16];
    v16bf bf1 = *(const v16bf*)&bs[(wn * 16 + lrow) * LSTR + 32 + lhalf * 16];
    v16bf a00 = *(const v16bf*)&as[(wm * 64 +  0 + lrow) * LSTR +  0 + lhalf * 16];
    v16bf a01 = *(const v16bf*)&as[(wm * 64 + 16 + lrow) * LSTR +  0 + lhalf * 16];
    v16bf a02 = *(const v16bf*)&as[(wm * 64 + 32 + lrow) * LSTR +  0 + lhalf * 16];
    v16bf a03 = *(const v16bf*)&as[(wm * 64 + 48 + lrow) * LSTR +  0 + lhalf * 16];
    v16bf a10 = *(const v16bf*)&as[(wm * 64 +  0 + lrow) * LSTR + 32 + lhalf * 16];
    v16bf a11 = *(const v16bf*)&as[(wm * 64 + 16 + lrow) * LSTR + 32 + lhalf * 16];
    v16bf a12 = *(const v16bf*)&as[(wm * 64 + 32 + lrow) * LSTR + 32 + lhalf * 16];
    v16bf a13 = *(const v16bf*)&as[(wm * 64 + 48 + lrow) * LSTR + 32 + lhalf * 16];
    acc0 = __builtin_amdgcn_wmma_f32_16x16x32_bf16(false, a00, false, bf0, (short)0, acc0, false, false);
    acc1 = __builtin_amdgcn_wmma_f32_16x16x32_bf16(false, a01, false, bf0, (short)0, acc1, false, false);
    acc2 = __builtin_amdgcn_wmma_f32_16x16x32_bf16(false, a02, false, bf0, (short)0, acc2, false, false);
    acc3 = __builtin_amdgcn_wmma_f32_16x16x32_bf16(false, a03, false, bf0, (short)0, acc3, false, false);
    acc0 = __builtin_amdgcn_wmma_f32_16x16x32_bf16(false, a10, false, bf1, (short)0, acc0, false, false);
    acc1 = __builtin_amdgcn_wmma_f32_16x16x32_bf16(false, a11, false, bf1, (short)0, acc1, false, false);
    acc2 = __builtin_amdgcn_wmma_f32_16x16x32_bf16(false, a12, false, bf1, (short)0, acc2, false, false);
    acc3 = __builtin_amdgcn_wmma_f32_16x16x32_bf16(false, a13, false, bf1, (short)0, acc3, false, false);

    wait_async0();      // residual async work for tile t+1 (overlapped with WMMAs)
    __syncthreads();
  }

  const int col = n0 + wn * 16 + lrow;
  v8f accs[4] = {acc0, acc1, acc2, acc3};
  for (int ms = 0; ms < 4; ++ms) {
    v8f a = accs[ms];
    for (int r = 0; r < 8; ++r) {
      int row = m0 + wm * 64 + ms * 16 + lhalf * 8 + r;
      float v = a[r];
      if (col < Nreal) {
        if (bias) v += bias[col];
        if (relu) v = fmaxf(v, 0.f);
      } else {
        v = padfill;
      }
      if (Cf) Cf[(size_t)row * N + col] = v;
      if (Cb) Cb[(size_t)row * N + col] = (bf16_t)v;
    }
  }
}

// ---------------- model_out normalize (f32 out + bf16 mirror) ----------------
__global__ __launch_bounds__(256)
void k_monorm(const float* __restrict__ mo, float* __restrict__ out_mo, bf16_t* __restrict__ mob) {
  int b = blockIdx.x;
  __shared__ float red[9];
  float vv[3]; float ss = 0.f;
  for (int q = 0; q < 3; ++q) { vv[q] = mo[(size_t)b * DD + threadIdx.x + q * 256]; ss += vv[q] * vv[q]; }
  ss = block_reduce_sum(ss, red);
  float inv = 1.f / fmaxf(sqrtf(ss), 1e-12f);
  for (int q = 0; q < 3; ++q) {
    float nv = vv[q] * inv;
    out_mo[(size_t)b * DD + threadIdx.x + q * 256] = nv;
    mob[(size_t)b * DD + threadIdx.x + q * 256] = (bf16_t)nv;
  }
}

// ---------------- exact top-80 per row over bf16 sims ----------------
// Two-level 256-bin radix histogram -> exact 16-bit threshold -> compaction
// -> 80 block-wide argmax selections (deterministic via index tiebreak).
__global__ __launch_bounds__(256)
void k_topk(const bf16_t* __restrict__ sims, int N, int* __restrict__ idx_out) {
  int b = blockIdx.x;
  const unsigned short* row = (const unsigned short*)(sims + (size_t)b * N);
  __shared__ unsigned hist[256];
  __shared__ unsigned hist2[256];
  __shared__ int s_thr1, s_above1, s_thr16;
  __shared__ unsigned cand_cnt;
  __shared__ unsigned ckey[2048];
  __shared__ int   cidx[2048];
  __shared__ unsigned bestk[8];
  __shared__ int   bestid[8];
  __shared__ int   bestslot[8];

  for (int i = threadIdx.x; i < 256; i += 256) { hist[i] = 0u; hist2[i] = 0u; }
  __syncthreads();
  for (int i = threadIdx.x; i < N; i += 256) {
    unsigned short x = row[i];
    unsigned k = (x & 0x8000u) ? ((unsigned)(~x) & 0xFFFFu) : (unsigned)(x | 0x8000u);
    atomicAdd(&hist[k >> 8], 1u);
  }
  __syncthreads();
  if (threadIdx.x == 0) {
    unsigned cum = 0; int t1 = 0;
    for (int by = 255; by >= 0; --by) {
      if (cum + hist[by] >= (unsigned)VK) { t1 = by; break; }
      cum += hist[by];
    }
    s_thr1 = t1; s_above1 = (int)cum;
  }
  __syncthreads();
  int t1 = s_thr1;
  for (int i = threadIdx.x; i < N; i += 256) {
    unsigned short x = row[i];
    unsigned k = (x & 0x8000u) ? ((unsigned)(~x) & 0xFFFFu) : (unsigned)(x | 0x8000u);
    if ((int)(k >> 8) == t1) atomicAdd(&hist2[k & 0xFFu], 1u);
  }
  __syncthreads();
  if (threadIdx.x == 0) {
    int needed = VK - s_above1;   // >= 1 by construction
    unsigned cum = 0; int t2 = 0;
    for (int by = 255; by >= 0; --by) {
      if ((int)(cum + hist2[by]) >= needed) { t2 = by; break; }
      cum += hist2[by];
    }
    s_thr16 = (t1 << 8) | t2;
    cand_cnt = 0u;
  }
  __syncthreads();
  unsigned thr = (unsigned)s_thr16;
  for (int i = threadIdx.x; i < N; i += 256) {
    unsigned short x = row[i];
    unsigned k = (x & 0x8000u) ? ((unsigned)(~x) & 0xFFFFu) : (unsigned)(x | 0x8000u);
    if (k >= thr) {
      unsigned p = atomicAdd(&cand_cnt, 1u);
      if (p < 2048u) { ckey[p] = k; cidx[p] = i; }
    }
  }
  __syncthreads();
  int cc = (int)cand_cnt; if (cc > 2048) cc = 2048;
  for (int sel = 0; sel < VK; ++sel) {
    unsigned mk = 0u; int mi = 0x7fffffff; int ms = -1;
    for (int i = threadIdx.x; i < cc; i += 256) {
      unsigned k = ckey[i]; int id = cidx[i];
      if (k > mk || (k == mk && id < mi)) { mk = k; mi = id; ms = i; }
    }
    for (int o = 16; o > 0; o >>= 1) {
      unsigned ok = __shfl_down(mk, o, 32);
      int oi = __shfl_down(mi, o, 32);
      int os = __shfl_down(ms, o, 32);
      if (ok > mk || (ok == mk && oi < mi)) { mk = ok; mi = oi; ms = os; }
    }
    int lane = threadIdx.x & 31, wid = threadIdx.x >> 5;
    if (lane == 0) { bestk[wid] = mk; bestid[wid] = mi; bestslot[wid] = ms; }
    __syncthreads();
    if (threadIdx.x == 0) {
      unsigned bk = bestk[0]; int bi = bestid[0]; int bs = bestslot[0];
      for (int w = 1; w < 8; ++w) {
        if (bestk[w] > bk || (bestk[w] == bk && bestid[w] < bi)) {
          bk = bestk[w]; bi = bestid[w]; bs = bestslot[w];
        }
      }
      idx_out[b * VK + sel] = (bs >= 0) ? bi : 0;
      if (bs >= 0) ckey[bs] = 0u;
    }
    __syncthreads();
  }
}

// ---------------- gather selected vocab rows (f32 renorm) + 26 cosine scores ----------------
__global__ __launch_bounds__(256)
void k_gather(const float* __restrict__ vocab, const int* __restrict__ idx,
              const float* __restrict__ board_n,
              float* __restrict__ wemb, float* __restrict__ scores,
              float* __restrict__ out_w0) {
  int b = blockIdx.y, v = blockIdx.x;
  int id = idx[b * VK + v];
  const float* src = vocab + (size_t)id * DD;
  __shared__ float red[9];
  float vv[3]; float ss = 0.f;
  for (int q = 0; q < 3; ++q) { vv[q] = src[threadIdx.x + q * 256]; ss += vv[q] * vv[q]; }
  ss = block_reduce_sum(ss, red);
  float inv = 1.f / fmaxf(sqrtf(ss), 1e-8f);
  float* dst = wemb + ((size_t)b * VK + v) * DD;
  for (int q = 0; q < 3; ++q) {
    int d = threadIdx.x + q * 256;
    float nv = vv[q] * inv;
    dst[d] = nv;
    if (v == 0) out_w0[(size_t)b * DD + d] = nv;
    vv[q] = nv;
  }
  for (int j = 0; j < 26; ++j) {
    const float* bn = board_n + ((size_t)b * 26 + j) * DD;
    float p = 0.f;
    for (int q = 0; q < 3; ++q) p += vv[q] * bn[threadIdx.x + q * 256];
    p = block_reduce_sum(p, red);
    if (threadIdx.x == 0) scores[((size_t)b * VK + v) * 26 + j] = p;
  }
}

// ---------------- rank rewards ----------------
__global__ void k_reward(const float* __restrict__ scores, float* __restrict__ tot) {
  int b = blockIdx.x, v = threadIdx.x;
  if (v < VK) {
    const float* s = scores + ((size_t)b * VK + v) * 26;
    float mx = -1e30f; int arg = 9;
    for (int j = 9; j < 26; ++j) { float x = s[j]; if (x > mx) { mx = x; arg = j; } }
    int nc = 0;
    for (int j = 0; j < 9; ++j) if (s[j] > mx) nc++;
    float sec = (arg < 18) ? 0.0f : ((arg < 25) ? 1.0f : -10.0f);
    tot[b * VK + v] = (float)nc + sec;
  }
}

// ---------------- top/bottom-40 mean-pool + normalize ----------------
__global__ __launch_bounds__(256)
void k_minmax(const float* __restrict__ tot, const float* __restrict__ wemb,
              float* __restrict__ out_mx, float* __restrict__ out_mn) {
  int b = blockIdx.x;
  __shared__ float wt[VK];
  __shared__ int selmax[VK / 2];
  __shared__ int selmin[VK / 2];
  __shared__ float red[9];
  if (threadIdx.x < VK) wt[threadIdx.x] = tot[b * VK + threadIdx.x];
  __syncthreads();
  if (threadIdx.x == 0) {
    float w2[VK];
    for (int i = 0; i < VK; ++i) w2[i] = wt[i];
    for (int s = 0; s < VK / 2; ++s) {
      int mi = 0; float mv = -1e30f;
      for (int i = 0; i < VK; ++i) if (w2[i] > mv) { mv = w2[i]; mi = i; }
      selmax[s] = mi; w2[mi] = -1e30f;
    }
    for (int i = 0; i < VK; ++i) w2[i] = wt[i];
    for (int s = 0; s < VK / 2; ++s) {
      int mi = 0; float mv = 1e30f;
      for (int i = 0; i < VK; ++i) if (w2[i] < mv) { mv = w2[i]; mi = i; }
      selmin[s] = mi; w2[mi] = 1e30f;
    }
  }
  __syncthreads();
  for (int pass = 0; pass < 2; ++pass) {
    const int* sel = pass ? selmin : selmax;
    float* dst = (pass ? out_mn : out_mx) + (size_t)b * DD;
    float acc[3] = {0.f, 0.f, 0.f};
    for (int s = 0; s < VK / 2; ++s) {
      const float* e = wemb + ((size_t)b * VK + sel[s]) * DD;
      for (int q = 0; q < 3; ++q) acc[q] += e[threadIdx.x + q * 256];
    }
    float ss = 0.f;
    for (int q = 0; q < 3; ++q) { acc[q] /= (float)(VK / 2); ss += acc[q] * acc[q]; }
    ss = block_reduce_sum(ss, red);
    float inv = 1.f / fmaxf(sqrtf(ss), 1e-12f);
    for (int q = 0; q < 3; ++q) dst[threadIdx.x + q * 256] = acc[q] * inv;
    __syncthreads();
  }
}

// ---------------- host-side orchestration ----------------
extern "C" void kernel_launch(void* const* d_in, const int* in_sizes, int n_in,
                              void* d_out, int out_size, void* d_ws, size_t ws_size,
                              hipStream_t stream) {
  (void)in_sizes; (void)n_in; (void)out_size; (void)ws_size;
  const float* pos   = (const float*)d_in[0];
  const float* neg   = (const float*)d_in[1];
  const float* neut  = (const float*)d_in[2];
  const float* assas = (const float*)d_in[3];
  const float* vocab = (const float*)d_in[4];
  const float* W1 = (const float*)d_in[5];  const float* b1 = (const float*)d_in[6];
  const float* W2 = (const float*)d_in[7];  const float* b2 = (const float*)d_in[8];
  const float* W3 = (const float*)d_in[9];  const float* b3 = (const float*)d_in[10];
  const float* W4 = (const float*)d_in[11]; const float* b4 = (const float*)d_in[12];

  char* ws = (char*)d_ws;
  size_t off = 0;
  auto alloc = [&](size_t bytes) -> void* {
    void* p = ws + off;
    off += (bytes + 255) & ~(size_t)255;
    return p;
  };

  float*  board_n = (float*) alloc((size_t)NB * 26 * DD * 4);
  bf16_t* xb      = (bf16_t*)alloc((size_t)NB * 3072 * 2);
  bf16_t* W1t     = (bf16_t*)alloc((size_t)2304 * 3072 * 2);
  bf16_t* W2t     = (bf16_t*)alloc((size_t)1728 * 2304 * 2);
  bf16_t* W3t     = (bf16_t*)alloc((size_t)1024 * 1728 * 2);
  bf16_t* W4t     = (bf16_t*)alloc((size_t)768 * 1024 * 2);
  bf16_t* h1      = (bf16_t*)alloc((size_t)NB * 2304 * 2);
  bf16_t* h2      = (bf16_t*)alloc((size_t)NB * 1728 * 2);
  bf16_t* h3      = (bf16_t*)alloc((size_t)NB * 1024 * 2);
  float*  moF     = (float*) alloc((size_t)NB * DD * 4);
  bf16_t* moB     = (bf16_t*)alloc((size_t)NB * DD * 2);
  bf16_t* vocB    = (bf16_t*)alloc((size_t)NVP * DD * 2);
  bf16_t* simsB   = (bf16_t*)alloc((size_t)NB * NVP * 2);
  int*    topi    = (int*)   alloc((size_t)NB * VK * 4);
  float*  wemb    = (float*) alloc((size_t)NB * VK * DD * 4);
  float*  scr     = (float*) alloc((size_t)NB * VK * 26 * 4);
  float*  tot     = (float*) alloc((size_t)NB * VK * 4);

  float* outp   = (float*)d_out;
  float* out_mo = outp;
  float* out_w0 = outp + (size_t)NB * DD;
  float* out_mx = outp + 2 * (size_t)NB * DD;
  float* out_mn = outp + 3 * (size_t)NB * DD;

  k_pool<<<NB, 256, 0, stream>>>(pos, neg, neut, assas, xb, board_n);
  k_wt_cvt<<<dim3(3072 / 32, 2304 / 32), 256, 0, stream>>>(W1, W1t, 3072, 2304, 3072, 2304);
  k_wt_cvt<<<dim3(2304 / 32, 1728 / 32), 256, 0, stream>>>(W2, W2t, 2304, 1700, 2304, 1728);
  k_wt_cvt<<<dim3(1728 / 32, 1024 / 32), 256, 0, stream>>>(W3, W3t, 1700, 1000, 1728, 1024);
  k_wt_cvt<<<dim3(1024 / 32, 768 / 32), 256, 0, stream>>>(W4, W4t, 1000, 768, 1024, 768);
  k_vocab<<<NVP, 256, 0, stream>>>(vocab, vocB, NV);

  gemm_tn_bf16<<<dim3(2304 / 64, 2), 256, 0, stream>>>(xb, W1t, b1, nullptr, h1, NB, 2304, 3072, 2304, 1, 0.f);
  gemm_tn_bf16<<<dim3(1728 / 64, 2), 256, 0, stream>>>(h1, W2t, b2, nullptr, h2, NB, 1728, 2304, 1700, 1, 0.f);
  gemm_tn_bf16<<<dim3(1024 / 64, 2), 256, 0, stream>>>(h2, W3t, b3, nullptr, h3, NB, 1024, 1728, 1000, 1, 0.f);
  gemm_tn_bf16<<<dim3(768 / 64, 2), 256, 0, stream>>>(h3, W4t, b4, moF, nullptr, NB, 768, 1024, 768, 0, 0.f);
  k_monorm<<<NB, 256, 0, stream>>>(moF, out_mo, moB);

  gemm_tn_bf16<<<dim3(NVP / 64, 2), 256, 0, stream>>>(moB, vocB, nullptr, nullptr, simsB, NB, NVP, DD, NV, 0, -1e30f);

  k_topk<<<NB, 256, 0, stream>>>(simsB, NVP, topi);
  k_gather<<<dim3(VK, NB), 256, 0, stream>>>(vocab, topi, board_n, wemb, scr, out_w0);
  k_reward<<<NB, 128, 0, stream>>>(scr, tot);
  k_minmax<<<NB, 256, 0, stream>>>(tot, wemb, out_mx, out_mn);
}
